// CARAFE_56805237457474
// MI455X (gfx1250) — matrix operand order repaired and checked by reading, not compiled, not run on hardware
//
#include <hip/hip_runtime.h>
#include <hip/hip_bf16.h>

typedef __attribute__((ext_vector_type(16))) _Float16 v16h;
typedef __attribute__((ext_vector_type(8)))  _Float16 v8h;
typedef __attribute__((ext_vector_type(8)))  float    v8f;
typedef __attribute__((ext_vector_type(4)))  unsigned int uint32x4;
typedef __attribute__((ext_vector_type(8)))  int      int32x8;
typedef __attribute__((ext_vector_type(4)))  int      int32x4;

#define B_     4
#define C_     256
#define HF     64
#define WF     64
#define HW     (HF*WF)          // 4096
#define COMP_  64
#define KUP2   25
#define S2     4
#define NENC   100              // KUP2*S2
#define NPAD   112              // 7 * 16
#define KENC   576              // COMP_*9, = 18*32
#define EPS_   1e-5f

// ---------------------------------------------------------------------------
// Workspace layout (bytes):
//   wch    f16 [64][256]               offset 0          size 32768
//   wet    f16 [112][576]              offset 32768      size 129024
//   act_t  f16 NHWC [4][4096][64]      offset 161792     size 2097152
//   logits f32 [(b*4+so)*25+ko][hw]    offset 2258944    size 6553600
//   mask   f32 same layout             offset 8812544    size 6553600
//   xh     f16 NHWC [4][4096][256]     offset 15366144   size 8388608
// total ~22.7 MB
// ---------------------------------------------------------------------------
#define WS_WCH    0
#define WS_WET    32768
#define WS_ACT    161792
#define WS_LOGITS 2258944
#define WS_MASK   8812544
#define WS_XH     15366144

// 16-bit A 16x32, contiguous k order used here: lane group 0-15 holds K {0..15},
// lanes 16-31 hold K {16..31}; element e -> K=kb+e
__device__ __forceinline__ int a_kb(int hi) { return hi ? 16 : 0; }

// ---------------------------------------------------------------------------
// k0a: pack weights to f16 GEMM-friendly layouts
// ---------------------------------------------------------------------------
__global__ void k0_pack(const float* __restrict__ w_comp,
                        const float* __restrict__ w_enc,
                        _Float16* __restrict__ wch,
                        _Float16* __restrict__ wet) {
    int i = blockIdx.x * blockDim.x + threadIdx.x;
    if (i < COMP_ * C_) {
        wch[i] = (_Float16)w_comp[i];               // [o][c] already B layout
    }
    int j = i - COMP_ * C_;
    if (j >= 0 && j < NPAD * KENC) {
        int n = j / KENC;
        int k = j % KENC;                           // k = tap*64 + c
        int tap = k >> 6;
        int c = k & 63;
        float v = 0.0f;
        if (n < NENC) v = w_enc[(size_t)(n * COMP_ + c) * 9 + tap];
        wet[j] = (_Float16)v;
    }
}

// ---------------------------------------------------------------------------
// k0b: NCHW f32 -> NHWC f16 transpose of x (LDS-tiled, 32x32)
// ---------------------------------------------------------------------------
__global__ void k0_transpose(const float* __restrict__ x,
                             _Float16* __restrict__ xh) {
    __shared__ _Float16 t[32][33];
    const int b  = blockIdx.x >> 10;                // 8 ctiles * 128 ptiles
    const int r  = blockIdx.x & 1023;
    const int c0 = (r >> 7) * 32;
    const int p0 = (r & 127) * 32;
    const int tid = threadIdx.x;
    const int hiq = tid >> 5;                       // 0..7
    const int lo  = tid & 31;
#pragma unroll
    for (int cc = hiq; cc < 32; cc += 8)
        t[cc][lo] = (_Float16)x[((size_t)(b * C_ + c0 + cc)) * HW + p0 + lo];
    __syncthreads();
#pragma unroll
    for (int pp = hiq; pp < 32; pp += 8)
        xh[((size_t)(b * HW) + p0 + pp) * C_ + c0 + lo] = t[lo][pp];
}

// ---------------------------------------------------------------------------
// k1: 1x1 compression GEMM (WMMA) + BN + SiLU -> act_t (f16, NHWC)
// B-panel (w_comp, 32KB f16) staged in LDS via TDM tensor_load_to_lds.
// one wave = 16-pixel M tile x N=64; K=256 in 8 steps of 32
// ---------------------------------------------------------------------------
__global__ void k1_compress(const _Float16* __restrict__ xh,
                            const _Float16* __restrict__ wch,
                            const float* __restrict__ gamma,
                            const float* __restrict__ beta,
                            const float* __restrict__ mean,
                            const float* __restrict__ var,
                            _Float16* __restrict__ act_t) {
    __shared__ _Float16 lwch[COMP_ * C_];           // 32 KB

    // --- TDM: DMA the whole f16 B panel into LDS (wave 0 issues) ---
    if ((threadIdx.x >> 5) == 0) {
        const uint64_t ga = (uint64_t)(uintptr_t)wch;
        const uint32_t lb = (uint32_t)(uintptr_t)(&lwch[0]);
        uint32x4 g0;
        g0[0] = 1u;                                               // count=1
        g0[1] = lb;                                               // lds_addr
        g0[2] = (uint32_t)ga;                                     // gaddr lo
        g0[3] = (uint32_t)((ga >> 32) & 0x01FFFFFFu) | (2u << 30);// gaddr hi|type=2
        int32x8 g1;
        g1[0] = (int)(1u << 16);          // data_size = 1 -> 2 bytes
        g1[1] = (int)(256u << 16);        // tensor_dim0 = 256 (lo16 @ bits 63:48)
        g1[2] = (int)(64u << 16);         // tensor_dim0 hi=0 | tensor_dim1=64 lo
        g1[3] = (int)(256u << 16);        // tensor_dim1 hi=0 | tile_dim0 = 256
        g1[4] = 64;                       // tile_dim1 = 64 | tile_dim2 = 0
        g1[5] = 256;                      // tensor_dim0_stride = 256 (lo32)
        g1[6] = 0;
        g1[7] = 0;
        const int32x4 z4 = {0, 0, 0, 0};
        const int32x8 z8 = {0, 0, 0, 0, 0, 0, 0, 0};
        __builtin_amdgcn_tensor_load_to_lds(g0, g1, z4, z4, z8, 0);
        __builtin_amdgcn_s_wait_tensorcnt(0);
    }
    __syncthreads();

    const int lane = threadIdx.x & 31;
    const int wid  = threadIdx.x >> 5;
    const int tile = blockIdx.x * 8 + wid;          // 0..1023
    const int hi   = lane >> 4;
    const int lm   = lane & 15;
    const int p_base = tile * 16;
    const int kb   = a_kb(hi);

    const int pA = p_base + lm;                     // this lane's A row (pixel)
    const _Float16* xp = xh + (size_t)pA * C_;

    v8f acc[4];
    const v8f vz = {0.f,0.f,0.f,0.f,0.f,0.f,0.f,0.f};
#pragma unroll
    for (int nt = 0; nt < 4; ++nt) acc[nt] = vz;

    for (int k0 = 0; k0 < 8; ++k0) {
        const int cb = k0 * 32;
        const v8h a0 = *(const v8h*)(xp + cb + kb);
        const v8h a1 = *(const v8h*)(xp + cb + 8 + kb);
        v16h a;
#pragma unroll
        for (int e = 0; e < 8; ++e) { a[e] = a0[e]; a[8 + e] = a1[e]; }
        const int kstart = cb + (hi ? 16 : 0);
#pragma unroll
        for (int nt = 0; nt < 4; ++nt) {
            v16h b = *(const v16h*)(lwch + (size_t)(nt * 16 + lm) * C_ + kstart);
            acc[nt] = __builtin_amdgcn_wmma_f32_16x16x32_f16(
                false, a, false, b, (short)0, acc[nt], false, false);
        }
    }

    // BN + SiLU epilogue, store f16 act in NHWC [pixel][comp]
#pragma unroll
    for (int nt = 0; nt < 4; ++nt) {
        const int o = nt * 16 + lm;                 // comp channel (per-lane N)
        const float inv = gamma[o] * rsqrtf(var[o] + EPS_);
        const float mu = mean[o];
        const float bb = beta[o];
#pragma unroll
        for (int v = 0; v < 8; ++v) {
            const int p = p_base + v + hi * 8;
            float t = (acc[nt][v] - mu) * inv + bb;
            float s = t * (1.0f / (1.0f + __expf(-t)));
            act_t[(size_t)p * COMP_ + o] = (_Float16)s;
        }
    }
}

// ---------------------------------------------------------------------------
// k2: 3x3 encoder conv as implicit WMMA GEMM -> logits
// A from NHWC f16 act (two b128 loads per K-step), B from packed f16 weights.
// M=16 pixels/wave, N=112 (7 tiles), K=576 (18 steps: 9 taps x 2 c-halves)
// ---------------------------------------------------------------------------
__global__ void k2_encoder(const _Float16* __restrict__ act_t,
                           const _Float16* __restrict__ wet,
                           float* __restrict__ logits) {
    const int lane = threadIdx.x & 31;
    const int wid  = threadIdx.x >> 5;
    const int tile = blockIdx.x * 8 + wid;          // 0..1023
    const int hi   = lane >> 4;
    const int lm   = lane & 15;
    const int p_base = tile * 16;
    const int kb   = a_kb(hi);

    const int pA  = p_base + lm;
    const int bA  = pA >> 12;
    const int hwA = pA & 4095;
    const int y0  = hwA >> 6;
    const int x0  = hwA & 63;

    v8f acc[7];
    const v8f vz = {0.f,0.f,0.f,0.f,0.f,0.f,0.f,0.f};
#pragma unroll
    for (int nt = 0; nt < 7; ++nt) acc[nt] = vz;

    for (int ks = 0; ks < 18; ++ks) {
        const int tap   = ks >> 1;
        const int chalf = ks & 1;
        const int ky = tap / 3;
        const int kx = tap % 3;
        const int y  = y0 + ky - 1;
        const int xx = x0 + kx - 1;
        const bool inb = ((unsigned)y < (unsigned)HF) && ((unsigned)xx < (unsigned)WF);

        v16h a;
        if (inb) {
            const _Float16* ap =
                act_t + ((size_t)(bA * HW + y * WF + xx)) * COMP_ + chalf * 32 + kb;
            const v8h a0 = *(const v8h*)(ap);
            const v8h a1 = *(const v8h*)(ap + 8);
#pragma unroll
            for (int e = 0; e < 8; ++e) { a[e] = a0[e]; a[8 + e] = a1[e]; }
        } else {
#pragma unroll
            for (int e = 0; e < 16; ++e) a[e] = (_Float16)0.0f;
        }

        const int kstart = ks * 32 + (hi ? 16 : 0);
#pragma unroll
        for (int nt = 0; nt < 7; ++nt) {
            v16h b = *(const v16h*)(wet + (size_t)(nt * 16 + lm) * KENC + kstart);
            acc[nt] = __builtin_amdgcn_wmma_f32_16x16x32_f16(
                false, a, false, b, (short)0, acc[nt], false, false);
        }
    }

    // store logits: layout ((b*4+so)*25+ko)*4096 + hw;  n = ko*4 + so
#pragma unroll
    for (int nt = 0; nt < 7; ++nt) {
        const int n = nt * 16 + lm;
        if (n < NENC) {
            const int ko = n >> 2;
            const int so = n & 3;
#pragma unroll
            for (int v = 0; v < 8; ++v) {
                const int p  = p_base + v + hi * 8;
                const int b_ = p >> 12;
                const int hw = p & 4095;
                logits[((size_t)((b_ * 4 + so) * KUP2 + ko)) * HW + hw] = acc[nt][v];
            }
        }
    }
}

// ---------------------------------------------------------------------------
// k3: softmax over the 25 taps for each (b, so, pixel)
// ---------------------------------------------------------------------------
__global__ void k3_softmax(const float* __restrict__ logits,
                           float* __restrict__ mask) {
    const int t = blockIdx.x * blockDim.x + threadIdx.x;   // 0..65535
    if (t >= B_ * S2 * HW) return;
    const int bso = t >> 12;                               // b*4+so
    const int hw  = t & 4095;
    const size_t base = (size_t)bso * KUP2 * HW + hw;

    float v[KUP2];
    float mx = -3.4e38f;
#pragma unroll
    for (int k = 0; k < KUP2; ++k) {
        v[k] = logits[base + (size_t)k * HW];
        mx = fmaxf(mx, v[k]);
    }
    float sum = 0.0f;
#pragma unroll
    for (int k = 0; k < KUP2; ++k) {
        v[k] = __expf(v[k] - mx);
        sum += v[k];
    }
    const float r = 1.0f / sum;
#pragma unroll
    for (int k = 0; k < KUP2; ++k) {
        mask[base + (size_t)k * HW] = v[k] * r;
    }
}

// ---------------------------------------------------------------------------
// k4: content-aware reassembly.  One block = (b, pixel); 256 threads = channels.
// Mask coefficients staged in LDS via async global->LDS copy (ASYNCcnt path).
// ---------------------------------------------------------------------------
__global__ void k4_reassemble(const float* __restrict__ x,
                              const float* __restrict__ mask,
                              float* __restrict__ out) {
    __shared__ float smask[NENC];
    const int b  = blockIdx.x >> 12;
    const int hw = blockIdx.x & 4095;
    const int i  = hw >> 6;
    const int j  = hw & 63;
    const int tid = threadIdx.x;

    if (tid < NENC) {
        // smask[so*25+ko] = mask[(b*100 + so*25+ko)*4096 + hw]
        const uint32_t lds_off = (uint32_t)(uintptr_t)(&smask[tid]);
        const uint64_t gaddr =
            (uint64_t)(uintptr_t)(mask + ((size_t)(b * NENC + tid)) * HW + hw);
        asm volatile("global_load_async_to_lds_b32 %0, %1, off"
                     :: "v"(lds_off), "v"(gaddr) : "memory");
    }
    asm volatile("s_wait_asynccnt 0x0" ::: "memory");
    __syncthreads();

    const int c = tid;
    const float* xc = x + (size_t)(b * C_ + c) * HW;

    float xa[KUP2];
#pragma unroll
    for (int t = 0; t < KUP2; ++t) {
        const int dy = t / 5;
        const int dx = t % 5;
        const int y  = i + dy - 2;
        const int xx = j + dx - 2;
        const bool inb = ((unsigned)y < (unsigned)HF) && ((unsigned)xx < (unsigned)WF);
        xa[t] = inb ? xc[y * WF + xx] : 0.0f;
    }

    float* oc = out + (size_t)(b * C_ + c) * (HW * 4);     // (128x128)
#pragma unroll
    for (int so = 0; so < S2; ++so) {
        const float* mrow = smask + so * KUP2;
        float sum = 0.0f;
#pragma unroll
        for (int t = 0; t < KUP2; ++t) sum = fmaf(xa[t], mrow[t], sum);
        const int s0 = so >> 1;
        const int s1 = so & 1;
        oc[(size_t)(i * 2 + s0) * (WF * 2) + (j * 2 + s1)] = sum;
    }
}

// ---------------------------------------------------------------------------
extern "C" void kernel_launch(void* const* d_in, const int* in_sizes, int n_in,
                              void* d_out, int out_size, void* d_ws, size_t ws_size,
                              hipStream_t stream) {
    const float* x       = (const float*)d_in[0];
    const float* w_comp  = (const float*)d_in[1];
    const float* gamma   = (const float*)d_in[2];
    const float* beta    = (const float*)d_in[3];
    const float* mean    = (const float*)d_in[4];
    const float* var     = (const float*)d_in[5];
    const float* w_enc   = (const float*)d_in[6];
    float* out = (float*)d_out;

    char* ws = (char*)d_ws;
    _Float16* wch    = (_Float16*)(ws + WS_WCH);
    _Float16* wet    = (_Float16*)(ws + WS_WET);
    _Float16* act_t  = (_Float16*)(ws + WS_ACT);
    float*    logits = (float*)(ws + WS_LOGITS);
    float*    mask   = (float*)(ws + WS_MASK);
    _Float16* xh     = (_Float16*)(ws + WS_XH);

    // k0a: pack weights
    {
        int total = COMP_ * C_ + NPAD * KENC;
        k0_pack<<<(total + 255) / 256, 256, 0, stream>>>(w_comp, w_enc, wch, wet);
    }
    // k0b: x -> NHWC f16
    k0_transpose<<<B_ * 8 * 128, 256, 0, stream>>>(x, xh);
    // k1: compression GEMM (TDM-staged B) + BN + SiLU
    k1_compress<<<128, 256, 0, stream>>>(xh, wch, gamma, beta, mean, var, act_t);
    // k2: encoder conv (implicit WMMA GEMM)
    k2_encoder<<<128, 256, 0, stream>>>(act_t, wet, logits);
    // k3: softmax over taps
    k3_softmax<<<(B_ * S2 * HW + 255) / 256, 256, 0, stream>>>(logits, mask);
    // k4: reassembly / upsample (async mask staging)
    k4_reassemble<<<B_ * HW, 256, 0, stream>>>(x, mask, out);
    (void)in_sizes; (void)n_in; (void)out_size; (void)ws_size;
}
